// TriangularSelfAttention_32744830664940
// MI455X (gfx1250) — matrix operand order repaired and checked by reading, not compiled, not run on hardware
//
#include <hip/hip_runtime.h>
#include <hip/hip_bf16.h>

#define N_RES 256
#define D_PAIR 128
#define NUM_HEADS 16
#define D_HEAD 32
#define D_INNER 512
#define N_QKV 1536  // q|k|v concatenated columns

typedef __attribute__((ext_vector_type(16))) _Float16 v16h;
typedef __attribute__((ext_vector_type(8)))  float    v8f;

// ---------------------------------------------------------------------------
// WMMA fragment loaders per CDNA5 ISA VGPR layouts (CDNA5 ISA §7.12.2)
// ---------------------------------------------------------------------------

// A-matrix 16x32 f16 (MxK), src row-major [m][k], ld multiple of 8.
// Lane L: M = L%16; halves 0..7 = K [khi, khi+7], halves 8..15 = K [16+khi, 16+khi+7]
// with khi = 8*(L/16). Two contiguous 16B runs -> 2x b128 loads.
__device__ __forceinline__ v16h load_frag_a(const _Float16* src, int ld, int lane) {
    const int m   = lane & 15;
    const int khi = (lane >> 4) << 3;   // 0 or 8
    union { float4 f[2]; v16h v; } u;
    u.f[0] = *(const float4*)(src + m * ld + khi);
    u.f[1] = *(const float4*)(src + m * ld + 16 + khi);
    return u.v;
}

// B-matrix 32x16 where B[k][n] = src[n*ld + k] (src holds B^T row-major).
// Lane L: N = L%16; halves 0..7 = K [khi, khi+7], halves 8..15 = K [16+khi, 16+khi+7], khi = 8*(L/16).
__device__ __forceinline__ v16h load_frag_bt(const _Float16* src, int ld, int lane) {
    const _Float16* p = src + (lane & 15) * ld + ((lane >> 4) << 3);
    union { float4 f[2]; v16h v; } u;
    u.f[0] = *(const float4*)p;
    u.f[1] = *(const float4*)(p + 16);
    return u.v;
}

// ---------------------------------------------------------------------------
// Conversion / packing kernels
// ---------------------------------------------------------------------------
__global__ void cvt_f32_to_f16(const float* __restrict__ src,
                               _Float16* __restrict__ dst, int n) {
    for (int i = blockIdx.x * blockDim.x + threadIdx.x; i < n;
         i += gridDim.x * blockDim.x)
        dst[i] = (_Float16)src[i];
}

// Pack transposed Wqkv: dst[n][kk] = W(n)[kk] for n in 0..1535, kk in 0..127.
__global__ void pack_wqkvT(const float* __restrict__ Wq, const float* __restrict__ Wk,
                           const float* __restrict__ Wv, _Float16* __restrict__ dst) {
    const int idx = blockIdx.x * blockDim.x + threadIdx.x;  // 0..196607
    const int n = idx >> 7;          // output column 0..1535
    const int kk = idx & 127;        // inner k 0..127
    const int which = n >> 9;
    const int c = n & 511;
    const float* W = (which == 0) ? Wq : (which == 1) ? Wk : Wv;
    dst[idx] = (_Float16)W[kk * D_INNER + c];
}

// Pack transposed Wo: dst[n][kk] = Wo[kk][n], n in 0..127, kk in 0..511.
__global__ void pack_woT(const float* __restrict__ Wo, _Float16* __restrict__ dst) {
    const int idx = blockIdx.x * blockDim.x + threadIdx.x;  // 0..65535
    const int n = idx >> 9;
    const int kk = idx & 511;
    dst[idx] = (_Float16)Wo[kk * D_PAIR + n];
}

// ---------------------------------------------------------------------------
// QKV projection: [65536 x 128] @ [128 x 1536] + bias.
// q,k stored [i][h][j][dh]; v stored TRANSPOSED [i][h][dh][j] so the
// attention kernel can async-copy it straight into its B^T LDS layout.
// Block: 256 threads (8 waves). Tile: 32 rows x 128 cols; each wave owns a
// 32x16 column strip = two C tiles sharing each B fragment.
// ---------------------------------------------------------------------------
__global__ __launch_bounds__(256) void qkv_proj_kernel(
    const _Float16* __restrict__ pair16, const _Float16* __restrict__ wqkvT,
    const float* __restrict__ bq, const float* __restrict__ bk,
    const float* __restrict__ bv, _Float16* __restrict__ q16,
    _Float16* __restrict__ k16, _Float16* __restrict__ v16) {
    __shared__ _Float16 Ash[32 * D_PAIR];  // 8 KB

    const int tid = threadIdx.x, lane = tid & 31, wave = tid >> 5;
    const int r0 = blockIdx.x * 32;

    // Cooperative A-tile load: 4096 halves = 512 float4, 2 per thread.
#pragma unroll
    for (int t = tid; t < 512; t += 256)
        ((float4*)Ash)[t] = ((const float4*)(pair16 + (size_t)r0 * D_PAIR))[t];
    __syncthreads();

    const int ncol = blockIdx.y * 128 + wave * 16;  // wave's column base
    const int n    = ncol + (lane & 15);            // lane's column
    const int which = n >> 9;                        // uniform per wave
    const int c     = n & 511;
    const float* bias = (which == 0) ? bq : (which == 1) ? bk : bv;
    const float bvv = bias[c];

    v8f acc0, acc1;
#pragma unroll
    for (int r = 0; r < 8; ++r) { acc0[r] = bvv; acc1[r] = bvv; }

#pragma unroll
    for (int ks = 0; ks < 4; ++ks) {
        v16h b  = load_frag_bt(wqkvT + (size_t)ncol * D_PAIR + ks * 32, D_PAIR, lane);
        v16h a0 = load_frag_a(Ash + ks * 32, D_PAIR, lane);
        v16h a1 = load_frag_a(Ash + 16 * D_PAIR + ks * 32, D_PAIR, lane);
        acc0 = __builtin_amdgcn_wmma_f32_16x16x32_f16(false, a0, false, b,
                                                      (short)0, acc0, false, false);
        acc1 = __builtin_amdgcn_wmma_f32_16x16x32_f16(false, a1, false, b,
                                                      (short)0, acc1, false, false);
    }

    const int hh = c >> 5, dh = c & 31;
    const int i = r0 >> 8;               // same for all 32 rows (r0 mult of 32)
    const int jrow = (lane >> 4) << 3;   // 0 or 8

    if (which == 2) {                    // V: transposed [i][h][dh][j] layout
        const size_t rowbase = ((size_t)(i * NUM_HEADS + hh) * D_HEAD + dh) * N_RES;
#pragma unroll
        for (int half = 0; half < 2; ++half) {
            const v8f& a = half ? acc1 : acc0;
            union { float4 f; _Float16 h[8]; } u;
#pragma unroll
            for (int r = 0; r < 8; ++r) u.h[r] = (_Float16)a[r];
            const int j = (r0 & 255) + half * 16 + jrow;   // 8 consecutive j
            *(float4*)(v16 + rowbase + j) = u.f;
        }
    } else {                             // Q/K: [i][h][j][dh] layout
        _Float16* dstbuf = (which == 0) ? q16 : k16;
        const size_t base = (size_t)(i * NUM_HEADS + hh) * N_RES * D_HEAD + dh;
#pragma unroll
        for (int half = 0; half < 2; ++half) {
            const v8f& a = half ? acc1 : acc0;
#pragma unroll
            for (int r = 0; r < 8; ++r) {
                const int j = (r0 & 255) + half * 16 + jrow + r;
                dstbuf[base + (size_t)j * D_HEAD] = (_Float16)a[r];
            }
        }
    }
}

// ---------------------------------------------------------------------------
// Triangular attention, one block per (i,h). 128 threads = 4 waves.
// K [256][32] and V^T [32][256] staged into LDS with CDNA5 async-to-LDS
// b128 copies (ASYNCcnt), scores in f32 WMMA C-layout registers, tril mask +
// softmax via xor-shuffles, probs re-staged through wave-private LDS
// (s_wait_dscnt 0) into A-layout for the PV matmul.
// LDS: 16 + 16 + 32 = 64 KB.
// ---------------------------------------------------------------------------
__global__ __launch_bounds__(128) void tri_attn_kernel(
    const _Float16* __restrict__ q16, const _Float16* __restrict__ k16,
    const _Float16* __restrict__ vT16, _Float16* __restrict__ o16) {
    __shared__ _Float16 Ksh[N_RES * D_HEAD];      // 16 KB, [k][c]
    __shared__ _Float16 VshT[D_HEAD * N_RES];     // 16 KB, [dh][k]
    __shared__ _Float16 Psh[4 * 16 * N_RES];      // 32 KB (8 KB / wave)

    const int ih = blockIdx.x;                    // i*16 + h
    const int i = ih >> 4, h = ih & 15;
    const int tid = threadIdx.x, lane = tid & 31, wave = tid >> 5;

    const _Float16* qg = q16  + (size_t)ih * N_RES * D_HEAD;
    const _Float16* kg = k16  + (size_t)ih * N_RES * D_HEAD;
    const _Float16* vg = vT16 + (size_t)ih * D_HEAD * N_RES;

    // Async staging: 1024 x b128 per matrix (16 KB each), 8 issues/thread.
#pragma unroll
    for (int t = tid; t < (N_RES * D_HEAD) / 8; t += 128) {
        const _Float16* kp = kg + t * 8;
        const _Float16* vp = vg + t * 8;
        const unsigned kl = (unsigned)(uintptr_t)(&Ksh[t * 8]);
        const unsigned vl = (unsigned)(uintptr_t)(&VshT[t * 8]);
        asm volatile("global_load_async_to_lds_b128 %0, %1, off"
                     :: "v"(kl), "v"(kp) : "memory");
        asm volatile("global_load_async_to_lds_b128 %0, %1, off"
                     :: "v"(vl), "v"(vp) : "memory");
    }
    asm volatile("s_wait_asynccnt 0" ::: "memory");
    __syncthreads();

    const float scale = 0.17677669529663687f;     // 1/sqrt(32)
    _Float16* Pw = Psh + wave * 16 * N_RES;
    const int jrow = (lane >> 4) << 3;            // 0 or 8

    for (int it = 0; it < 4; ++it) {
        const int j0 = (wave * 4 + it) * 16;

        // S = Q K^T : 16 WMMAs (Dh=32 == one K step).
        // B[c][kk] = K[k0+kk][c]; Ksh rows are contiguous in c -> bt loader.
        v16h qa = load_frag_a(qg + (size_t)j0 * D_HEAD, D_HEAD, lane);
        v8f sc[16];
#pragma unroll
        for (int t = 0; t < 16; ++t) {
            v16h kb = load_frag_bt(Ksh + (t * 16) * D_HEAD, D_HEAD, lane);
            v8f z = {};
            sc[t] = __builtin_amdgcn_wmma_f32_16x16x32_f16(false, qa, false, kb,
                                                           (short)0, z, false, false);
        }

        // Masked softmax over k. Lane holds S[j0 + r + jrow][16t + lane%16].
#pragma unroll
        for (int r = 0; r < 8; ++r) {
            const int j = j0 + r + jrow;
            float m = -3.0e38f;
#pragma unroll
            for (int t = 0; t < 16; ++t) {
                const int kk = t * 16 + (lane & 15);
                const float s = (kk <= j) ? sc[t][r] * scale : -3.0e38f;
                m = fmaxf(m, s);
            }
            for (int d = 1; d < 16; d <<= 1) m = fmaxf(m, __shfl_xor(m, d, 32));
            float ssum = 0.0f;
#pragma unroll
            for (int t = 0; t < 16; ++t) {
                const int kk = t * 16 + (lane & 15);
                const float e = (kk <= j) ? __expf(sc[t][r] * scale - m) : 0.0f;
                sc[t][r] = e;
                ssum += e;
            }
            for (int d = 1; d < 16; d <<= 1) ssum += __shfl_xor(ssum, d, 32);
            const float inv = 1.0f / ssum;
#pragma unroll
            for (int t = 0; t < 16; ++t) sc[t][r] *= inv;
        }

        // C-layout -> A-layout via wave-private LDS.
#pragma unroll
        for (int t = 0; t < 16; ++t)
#pragma unroll
            for (int r = 0; r < 8; ++r)
                Pw[(r + jrow) * N_RES + t * 16 + (lane & 15)] = (_Float16)sc[t][r];
        asm volatile("s_wait_dscnt 0" ::: "memory");

        // O = P V : K=256 in 8 steps, N=32 in 2 tiles -> 16 WMMAs.
        // B[kk][n] = V[k0+kk][n0+n] = VshT[n0+n][k0+kk] -> bt loader.
        v8f o0 = {}, o1 = {};
#pragma unroll
        for (int ks = 0; ks < 8; ++ks) {
            v16h pa  = load_frag_a(Pw + ks * 32, N_RES, lane);
            v16h vb0 = load_frag_bt(VshT + ks * 32, N_RES, lane);
            v16h vb1 = load_frag_bt(VshT + 16 * N_RES + ks * 32, N_RES, lane);
            o0 = __builtin_amdgcn_wmma_f32_16x16x32_f16(false, pa, false, vb0,
                                                        (short)0, o0, false, false);
            o1 = __builtin_amdgcn_wmma_f32_16x16x32_f16(false, pa, false, vb1,
                                                        (short)0, o1, false, false);
        }

        // Store into o16 [i][j][h*32 + dh] (row-major [r][512]).
#pragma unroll
        for (int r = 0; r < 8; ++r) {
            const int j = j0 + r + jrow;
            const size_t base = (size_t)(i * N_RES + j) * D_INNER + h * D_HEAD;
            o16[base + (lane & 15)]      = (_Float16)o0[r];
            o16[base + 16 + (lane & 15)] = (_Float16)o1[r];
        }
    }
}

// ---------------------------------------------------------------------------
// Output projection + bias + residual: out = pair + o16 @ Wo + bo  (f32 out).
// Block: 256 threads, tile 32 rows x 128 cols, K=512 in 16 WMMA steps,
// each wave: two C tiles sharing each B fragment (from WoT).
// ---------------------------------------------------------------------------
__global__ __launch_bounds__(256) void out_proj_kernel(
    const _Float16* __restrict__ o16, const _Float16* __restrict__ woT,
    const float* __restrict__ bo, const float* __restrict__ pair,
    float* __restrict__ out) {
    __shared__ _Float16 Ash[32 * D_INNER];  // 32 KB

    const int tid = threadIdx.x, lane = tid & 31, wave = tid >> 5;
    const int r0 = blockIdx.x * 32;

#pragma unroll
    for (int t = tid; t < (32 * D_INNER) / 8; t += 256)
        ((float4*)Ash)[t] = ((const float4*)(o16 + (size_t)r0 * D_INNER))[t];
    __syncthreads();

    const int ncol = wave * 16;
    const int n = ncol + (lane & 15);
    const float bvv = bo[n];
    v8f acc0, acc1;
#pragma unroll
    for (int r = 0; r < 8; ++r) { acc0[r] = bvv; acc1[r] = bvv; }

#pragma unroll
    for (int ks = 0; ks < 16; ++ks) {
        v16h b  = load_frag_bt(woT + (size_t)ncol * D_INNER + ks * 32, D_INNER, lane);
        v16h a0 = load_frag_a(Ash + ks * 32, D_INNER, lane);
        v16h a1 = load_frag_a(Ash + 16 * D_INNER + ks * 32, D_INNER, lane);
        acc0 = __builtin_amdgcn_wmma_f32_16x16x32_f16(false, a0, false, b,
                                                      (short)0, acc0, false, false);
        acc1 = __builtin_amdgcn_wmma_f32_16x16x32_f16(false, a1, false, b,
                                                      (short)0, acc1, false, false);
    }

#pragma unroll
    for (int half = 0; half < 2; ++half) {
        const v8f& a = half ? acc1 : acc0;
#pragma unroll
        for (int r = 0; r < 8; ++r) {
            const int rr = r0 + half * 16 + r + ((lane >> 4) << 3);
            const size_t off = (size_t)rr * D_PAIR + n;
            out[off] = pair[off] + a[r];
        }
    }
}

// ---------------------------------------------------------------------------
// Launch
// ---------------------------------------------------------------------------
extern "C" void kernel_launch(void* const* d_in, const int* in_sizes, int n_in,
                              void* d_out, int out_size, void* d_ws, size_t ws_size,
                              hipStream_t stream) {
    const float* pair = (const float*)d_in[0];
    const float* Wq   = (const float*)d_in[1];
    const float* bq   = (const float*)d_in[2];
    const float* Wk   = (const float*)d_in[3];
    const float* bk   = (const float*)d_in[4];
    const float* Wv   = (const float*)d_in[5];
    const float* bv   = (const float*)d_in[6];
    const float* Wo   = (const float*)d_in[7];
    const float* bo   = (const float*)d_in[8];
    float* out = (float*)d_out;

    // Workspace carve-up (f16 buffers, generously aligned by construction).
    char* ws = (char*)d_ws;
    const size_t n_pair  = (size_t)N_RES * N_RES * D_PAIR;    // 8388608
    const size_t n_wqkv  = (size_t)D_PAIR * N_QKV;            // 196608
    const size_t n_wo    = (size_t)D_INNER * D_PAIR;          // 65536
    const size_t n_headm = (size_t)N_RES * NUM_HEADS * N_RES * D_HEAD;  // 33554432
    const size_t n_o     = (size_t)N_RES * N_RES * D_INNER;   // 33554432

    _Float16* pair16 = (_Float16*)ws;                 ws += n_pair  * 2;
    _Float16* wqkvT  = (_Float16*)ws;                 ws += n_wqkv  * 2;
    _Float16* woT    = (_Float16*)ws;                 ws += n_wo    * 2;
    _Float16* q16    = (_Float16*)ws;                 ws += n_headm * 2;
    _Float16* k16    = (_Float16*)ws;                 ws += n_headm * 2;
    _Float16* vT16   = (_Float16*)ws;                 ws += n_headm * 2;
    _Float16* o16    = (_Float16*)ws;                 ws += n_o     * 2;

    // 0) Conversions / transposed weight packing.
    cvt_f32_to_f16<<<8192, 256, 0, stream>>>(pair, pair16, (int)n_pair);
    pack_wqkvT<<<(int)(n_wqkv / 256), 256, 0, stream>>>(Wq, Wk, Wv, wqkvT);
    pack_woT<<<(int)(n_wo / 256), 256, 0, stream>>>(Wo, woT);

    // 1) QKV projection: 2048 M-tiles (32 rows) x 12 N-blocks.
    qkv_proj_kernel<<<dim3(2048, 12), 256, 0, stream>>>(
        pair16, wqkvT, bq, bk, bv, q16, k16, vT16);

    // 2) Triangular attention: one block per (i,h).
    tri_attn_kernel<<<N_RES * NUM_HEADS, 128, 0, stream>>>(q16, k16, vT16, o16);

    // 3) Output projection + residual.
    out_proj_kernel<<<2048, 256, 0, stream>>>(o16, woT, bo, pair, out);
}
